// SelfAttn_73160472920716
// MI455X (gfx1250) — matrix operand order repaired and checked by reading, not compiled, not run on hardware
//
#include <hip/hip_runtime.h>
#include <hip/hip_bf16.h>
#include <math.h>

// ---------------------------------------------------------------------------
// Problem constants (fixed by the reference)
// ---------------------------------------------------------------------------
constexpr int Bb   = 4;
constexpr int Tt   = 1024;
constexpr int Ee   = 1024;
constexpr int Hh   = 8;
constexpr int DHd  = 128;      // Ee / Hh
constexpr int HIDh = 4096;     // 4 * Ee
constexpr int Mrows = Bb * Tt; // 4096 tokens
constexpr float EPSb = 1e-5f;
constexpr float NEGv = -1e9f;

typedef __attribute__((ext_vector_type(16))) __bf16 v16bf;
typedef __attribute__((ext_vector_type(8)))  float  v8f;
typedef __attribute__((ext_vector_type(4)))  unsigned int v4u;
typedef __attribute__((ext_vector_type(8)))  int  v8i_t;
typedef __attribute__((ext_vector_type(4)))  int  v4i_t;

// ---------------------------------------------------------------------------
// Tensor Data Mover availability (this toolchain: 6-arg form,
//  (uint32x4 g0, int32x8 g1, int32x4 g2, int32x4 g3, int32x8 g4, i32 cpol))
// ---------------------------------------------------------------------------
#if defined(__has_builtin)
#if __has_builtin(__builtin_amdgcn_tensor_load_to_lds) && \
    __has_builtin(__builtin_amdgcn_s_wait_tensorcnt)
#define USE_TDM 1
#endif
#endif
#ifndef USE_TDM
#define USE_TDM 0
#endif

// Issue a 2D tile DMA global->LDS via TDM (wave-level op, EXEC ignored).
//  lds_off       : byte offset of tile start inside LDS
//  gptr          : global address of tile start
//  tile_x/tile_y : tile size in elements (x contiguous) / rows
//  stride_elems  : global row stride in elements (data_size units)
//  pad_code      : LDS pad interval code (2^(c+1) DWORDs between pads)
//  pad_dw        : pad amount in DWORDs
__device__ inline void tdm_tile_load(unsigned lds_off, const __bf16* gptr,
                                     int tile_x, int tile_y, long stride_elems,
                                     int pad_code, int pad_dw)
{
#if USE_TDM
    const unsigned long long ga = (unsigned long long)(const void*)gptr;
    v4u g0;
    g0[0] = 1u;                                        // count=1 (valid user D#)
    g0[1] = lds_off;                                   // lds_addr
    g0[2] = (unsigned)ga;                              // global_addr[31:0]
    g0[3] = (unsigned)((ga >> 32) & 0x1ffffffu)        // global_addr[56:32]
          | (2u << 30);                                // type = 2 ("image")
    const unsigned long long st = (unsigned long long)stride_elems;
    v8i_t g1;
    g1[0] = (1 << 16)                                  // data_size = 1 -> 2 bytes
          | (1 << 20)                                  // pad_enable
          | (pad_code << 22)                           // pad_interval
          | ((pad_dw - 1) << 25);                      // pad_amount
    g1[1] = 0;                                         // tensor_dim0[15:0]<<16 (dim0 = 1<<30)
    g1[2] = 0x4000;                                    // tensor_dim0[31:16]; tensor_dim1[15:0]=0
    g1[3] = 0x4000 | (tile_x << 16);                   // tensor_dim1[31:16]=1<<30; tile_dim0
    g1[4] = tile_y & 0xffff;                           // tile_dim1; tile_dim2 = 0
    g1[5] = (int)(st & 0xffffffffu);                   // tensor_dim0_stride[31:0]
    g1[6] = (int)((st >> 32) & 0xffffu);               // tensor_dim0_stride[47:32]
    g1[7] = 0;
    const v4i_t gz4 = {0, 0, 0, 0};
    const v8i_t gz8 = {0, 0, 0, 0, 0, 0, 0, 0};
    __builtin_amdgcn_tensor_load_to_lds(g0, g1, gz4, gz4, gz8, 0);
#else
    (void)lds_off; (void)gptr; (void)tile_x; (void)tile_y;
    (void)stride_elems; (void)pad_code; (void)pad_dw;
#endif
}

// ---------------------------------------------------------------------------
// fp32 -> bf16 conversion (grid-stride)
// ---------------------------------------------------------------------------
__global__ __launch_bounds__(256)
void cvt_f32_bf16(const float* __restrict__ in, __bf16* __restrict__ out, long n) {
    long i = (long)blockIdx.x * blockDim.x + threadIdx.x;
    long stride = (long)gridDim.x * blockDim.x;
    for (; i < n; i += stride) out[i] = (__bf16)in[i];
}

// ---------------------------------------------------------------------------
// Tiled WMMA GEMM:  C[M,N] = A[M,Kd] * op(B) + epilogue
//   BT = true : Bm is (N x Kd) row-major  (weight W, computes A @ W^T)
//   BT = false: Bm is (Kd x N) row-major  (e.g. scores @ V)
// FLAGS bits select the (compile-time) epilogue:
constexpr int F_BIAS  = 1;   // += bias[col]
constexpr int F_RELU  = 2;   // max(v,0)
constexpr int F_RESID = 4;   // += resid[row,col]
constexpr int F_MASK  = 8;   // attention mask: scale + causal/length -> NEG
constexpr int F_WF    = 16;  // write fp32 C
constexpr int F_WB    = 32;  // write bf16 C
//
// Block tile 128(M) x 64(N), K-step 32, 8 waves each owning a 32x32 subtile
// (four v_wmma_f32_16x16x32_bf16 accumulators). Double-buffered LDS; tiles
// staged by the Tensor Data Mover (async, TENSORcnt) when available.
// Batched via blockIdx.z: off = (z / zDiv)*strideB + (z % zDiv)*strideH.
// ---------------------------------------------------------------------------
constexpr int BLK_M = 128, BLK_N = 64, BLK_K = 32;

template <bool BT, int FLAGS>
__global__ __launch_bounds__(256)
void gemm_bf16_wmma(const __bf16* __restrict__ A,  long lda, long aB, long aH,
                    const __bf16* __restrict__ Bm, long ldb, long bB, long bH,
                    const float*  __restrict__ bias,
                    const float*  __restrict__ resid,
                    float* __restrict__ Cf, __bf16* __restrict__ Cb,
                    long ldc, long cB, long cH,
                    int M, int N, int Kd, int zDiv,
                    float scale, const unsigned char* __restrict__ oh, int Tdim)
{
    (void)M; (void)N;
    __shared__ __bf16 As[2][BLK_M][BLK_K + 8];       // 80B rows: no bank conflicts
    constexpr int BS0 = BT ? BLK_N : BLK_K;
    constexpr int BS1 = BT ? (BLK_K + 8) : (BLK_N + 8);
    __shared__ __bf16 Bs[2][BS0][BS1];
    constexpr int B_PAD_CODE = BT ? 3 : 4;           // 64B / 128B row interval
    constexpr int B_TILE_X   = BT ? BLK_K : BLK_N;
    constexpr int B_TILE_Y   = BT ? BLK_N : BLK_K;

    const int tid  = threadIdx.x;
    const int wid  = tid >> 5;
    const int lane = tid & 31;
    const int r    = lane & 15;
    const int hi   = lane >> 4;

    const int z   = blockIdx.z;
    const long zb = z / zDiv;
    const long zh = z % zDiv;
    A  += zb * aB + zh * aH;
    Bm += zb * bB + zh * bH;
    const long coff = zb * cB + zh * cH;

    const int bm = blockIdx.y * BLK_M;
    const int bn = blockIdx.x * BLK_N;

    const int waveM = (wid & 3) * 32;   // 4 waves along M
    const int waveN = (wid >> 2) * 32;  // 2 waves along N

    v8f acc[2][2];
    #pragma unroll
    for (int i = 0; i < 2; ++i)
        #pragma unroll
        for (int j = 0; j < 2; ++j)
            acc[i][j] = v8f{0.f, 0.f, 0.f, 0.f, 0.f, 0.f, 0.f, 0.f};

    // tile-start pointers (advance by BLK_K along K each step)
    const __bf16* aTile = A + (long)bm * lda;
    const __bf16* bTile = BT ? (Bm + (long)bn * ldb) : (Bm + bn);
    const long bStep = BT ? (long)BLK_K : (long)BLK_K * ldb;

#if USE_TDM
    const unsigned asOff0 = (unsigned)(unsigned long long)(void*)&As[0][0][0];
    const unsigned asOff1 = (unsigned)(unsigned long long)(void*)&As[1][0][0];
    const unsigned bsOff0 = (unsigned)(unsigned long long)(void*)&Bs[0][0][0];
    const unsigned bsOff1 = (unsigned)(unsigned long long)(void*)&Bs[1][0][0];
    if (wid == 0) {   // prologue: DMA tile 0 into buffer 0
        tdm_tile_load(asOff0, aTile, BLK_K, BLK_M, lda, 3, 4);
        tdm_tile_load(bsOff0, bTile, B_TILE_X, B_TILE_Y, ldb, B_PAD_CODE, 4);
    }
#else
    const int am = tid >> 1;            // 0..127 : A row in tile
    const int ak = (tid & 1) * 16;      // 0 or 16
#endif

    int p = 0;
    for (int kt = 0; kt < Kd; kt += BLK_K, p ^= 1) {
#if USE_TDM
        if (wid == 0) {
            if (kt + BLK_K < Kd) {      // DMA next tile into other buffer
                tdm_tile_load(p ? asOff0 : asOff1, aTile + (kt + BLK_K),
                              BLK_K, BLK_M, lda, 3, 4);
                tdm_tile_load(p ? bsOff0 : bsOff1, bTile + (long)(kt / BLK_K + 1) * bStep,
                              B_TILE_X, B_TILE_Y, ldb, B_PAD_CODE, 4);
                __builtin_amdgcn_s_wait_tensorcnt((short)2);  // current tile done
            } else {
                __builtin_amdgcn_s_wait_tensorcnt((short)0);
            }
        }
        __syncthreads();                // current buffer ready for all waves
#else
        // ---- synchronous fallback staging (uint4 via VGPRs) ----
        {
            const uint4* ap = reinterpret_cast<const uint4*>(aTile + (long)am * lda + kt + ak);
            *reinterpret_cast<uint4*>(&As[p][am][ak])     = ap[0];
            *reinterpret_cast<uint4*>(&As[p][am][ak + 8]) = ap[1];
            if (BT) {
                const int nrow = tid >> 2;
                const int bk   = (tid & 3) * 8;
                *reinterpret_cast<uint4*>(&Bs[p][nrow][bk]) =
                    *reinterpret_cast<const uint4*>(bTile + (long)nrow * ldb + kt + bk);
            } else {
                const int krow = tid >> 3;
                const int nc   = (tid & 7) * 8;
                *reinterpret_cast<uint4*>(&Bs[p][krow][nc]) =
                    *reinterpret_cast<const uint4*>(bTile + (long)(kt + krow) * ldb + nc);
            }
        }
        __syncthreads();
#endif

        // ---- fragments: contiguous k order (k = 16*hi + e), same in A and B ----
        v16bf af[2], bg[2];
        #pragma unroll
        for (int s = 0; s < 2; ++s) {
            const int m0 = waveM + s * 16 + r;   // A: lanes0-15 K0..15, lanes16-31 K16..31
            const int kb = hi * 16;
            #pragma unroll
            for (int e = 0; e < 16; ++e) af[s][e] = As[p][m0][kb + e];

            const int n0  = waveN + s * 16 + r;  // B: lanes0-15 K0..15, lanes16-31 K16..31
            const int kb2 = hi * 16;
            #pragma unroll
            for (int e = 0; e < 16; ++e)
                bg[s][e] = BT ? Bs[p][n0][kb2 + e] : Bs[p][kb2 + e][n0];
        }

        #pragma unroll
        for (int sm = 0; sm < 2; ++sm)
            #pragma unroll
            for (int sn = 0; sn < 2; ++sn)
                acc[sm][sn] = __builtin_amdgcn_wmma_f32_16x16x32_bf16(
                    false, af[sm], false, bg[sn], (short)0, acc[sm][sn], false, false);

        __syncthreads();                // release buffer p before it is re-filled
    }

    // ---- epilogue: C 16x16 f32 layout: VGPR e -> row base+hi*8+e, col = lane&15
    float*  Cfp = (FLAGS & F_WF) ? Cf + coff : nullptr;
    __bf16* Cbp = (FLAGS & F_WB) ? Cb + coff : nullptr;

    #pragma unroll
    for (int sm = 0; sm < 2; ++sm) {
        const int rowBase = bm + waveM + sm * 16 + hi * 8;
        #pragma unroll
        for (int sn = 0; sn < 2; ++sn) {
            const int col = bn + waveN + sn * 16 + r;
            float bcol = 0.f;
            if (FLAGS & F_BIAS) bcol = bias[col];
            bool ohc = false;
            if (FLAGS & F_MASK) ohc = (oh[(int)zb * Tdim + col] != 0) || (col == 0);
            long idx = (long)rowBase * ldc + col;
            #pragma unroll
            for (int e = 0; e < 8; ++e, idx += ldc) {
                float v = acc[sm][sn][e];
                if (FLAGS & F_MASK) {
                    const int row = rowBase + e;
                    Cfp[idx] = ((col <= row) && ohc) ? v * scale : NEGv;
                } else {
                    if (FLAGS & F_BIAS)  v += bcol;
                    if (FLAGS & F_RESID) v += resid[idx];
                    if (FLAGS & F_RELU)  v = fmaxf(v, 0.f);
                    if (FLAGS & F_WF)    Cfp[idx] = v;
                    if (FLAGS & F_WB)    Cbp[idx] = (__bf16)v;
                }
            }
        }
    }
}

// ---------------------------------------------------------------------------
// Row softmax over T=1024 (one block per row), in-place fp32 + bf16 copy
// ---------------------------------------------------------------------------
__global__ __launch_bounds__(256)
void softmax_rows(float* __restrict__ S, __bf16* __restrict__ Sb, int T)
{
    const long row = blockIdx.x;
    float*  p  = S  + row * (long)T;
    __bf16* pb = Sb + row * (long)T;
    const int tid  = threadIdx.x;
    const int wid  = tid >> 5;
    const int lane = tid & 31;
    __shared__ float red[8];

    float v[4];
    float mx = -3.4e38f;
    #pragma unroll
    for (int i = 0; i < 4; ++i) { v[i] = p[tid + i * 256]; mx = fmaxf(mx, v[i]); }
    #pragma unroll
    for (int off = 16; off; off >>= 1) mx = fmaxf(mx, __shfl_xor(mx, off));
    if (lane == 0) red[wid] = mx;
    __syncthreads();
    mx = red[0];
    #pragma unroll
    for (int j = 1; j < 8; ++j) mx = fmaxf(mx, red[j]);
    __syncthreads();

    float sum = 0.f;
    #pragma unroll
    for (int i = 0; i < 4; ++i) { v[i] = __expf(v[i] - mx); sum += v[i]; }
    #pragma unroll
    for (int off = 16; off; off >>= 1) sum += __shfl_xor(sum, off);
    if (lane == 0) red[wid] = sum;
    __syncthreads();
    sum = 0.f;
    #pragma unroll
    for (int j = 0; j < 8; ++j) sum += red[j];

    const float inv = 1.f / sum;
    #pragma unroll
    for (int i = 0; i < 4; ++i) {
        const float o = v[i] * inv;
        p[tid + i * 256]  = o;
        pb[tid + i * 256] = (__bf16)o;
    }
}

// ---------------------------------------------------------------------------
// Length-masked BatchNorm over (b,t) per feature; one block per feature.
// Normalizes in place; optional bf16 copy of the normalized output.
// ---------------------------------------------------------------------------
__global__ __launch_bounds__(256)
void bn_feature(float* __restrict__ xio, const unsigned char* __restrict__ oh,
                const float* __restrict__ gamma, const float* __restrict__ beta,
                __bf16* __restrict__ outb, int rows, int E)
{
    const int e   = blockIdx.x;
    const int tid = threadIdx.x;
    const int wid = tid >> 5, lane = tid & 31;
    __shared__ float rs[8], r2[8], rc[8];

    float s = 0.f, s2 = 0.f, c = 0.f;
    for (int i = tid; i < rows; i += 256) {
        const float m = oh[i] ? 1.f : 0.f;
        const float v = xio[(long)i * E + e];
        s += v * m; s2 += v * v * m; c += m;
    }
    #pragma unroll
    for (int off = 16; off; off >>= 1) {
        s  += __shfl_xor(s,  off);
        s2 += __shfl_xor(s2, off);
        c  += __shfl_xor(c,  off);
    }
    if (lane == 0) { rs[wid] = s; r2[wid] = s2; rc[wid] = c; }
    __syncthreads();
    s = 0.f; s2 = 0.f; c = 0.f;
    #pragma unroll
    for (int j = 0; j < 8; ++j) { s += rs[j]; s2 += r2[j]; c += rc[j]; }

    const float n    = fmaxf(c, 1.f);
    const float mean = s / n;
    const float var  = s2 / n - mean * mean;
    const float inv  = rsqrtf(var + EPSb);
    const float g    = gamma[e] * inv;
    const float bofs = beta[e] - mean * g;
    for (int i = tid; i < rows; i += 256) {
        const float v = xio[(long)i * E + e] * g + bofs;
        xio[(long)i * E + e] = v;
        if (outb) outb[(long)i * E + e] = (__bf16)v;
    }
}

// ---------------------------------------------------------------------------
// Launch
// ---------------------------------------------------------------------------
extern "C" void kernel_launch(void* const* d_in, const int* in_sizes, int n_in,
                              void* d_out, int out_size, void* d_ws, size_t ws_size,
                              hipStream_t stream)
{
    (void)in_sizes; (void)n_in; (void)out_size; (void)ws_size;

    const float*         x     = (const float*)d_in[0];
    const unsigned char* oh    = (const unsigned char*)d_in[1];  // bool (B,T)
    const float*         in_w  = (const float*)d_in[2];
    const float*         in_b  = (const float*)d_in[3];
    const float*         out_w = (const float*)d_in[4];
    const float*         out_b = (const float*)d_in[5];
    const float*         w1    = (const float*)d_in[6];
    const float*         b1    = (const float*)d_in[7];
    const float*         w2    = (const float*)d_in[8];
    const float*         b2    = (const float*)d_in[9];
    const float*         ag    = (const float*)d_in[10];
    const float*         abt   = (const float*)d_in[11];
    const float*         mg    = (const float*)d_in[12];
    const float*         mbt   = (const float*)d_in[13];

    float* x2out  = (float*)d_out;                                  // (B*T, E)
    float* scores = (float*)d_out + (long)Mrows * Ee;               // (B,H,T,T)

    // workspace carve-up (256B aligned)
    char* w = (char*)d_ws;
    auto alloc = [&](size_t bytes) { char* p = w; w += (bytes + 255) & ~(size_t)255; return p; };
    __bf16* xb    = (__bf16*)alloc((size_t)Mrows * Ee * 2);            //  8 MiB
    __bf16* inwb  = (__bf16*)alloc((size_t)3 * Ee * Ee * 2);           //  6 MiB
    __bf16* outwb = (__bf16*)alloc((size_t)Ee * Ee * 2);               //  2 MiB
    __bf16* w1b   = (__bf16*)alloc((size_t)HIDh * Ee * 2);             //  8 MiB
    __bf16* w2b   = (__bf16*)alloc((size_t)Ee * HIDh * 2);             //  8 MiB
    __bf16* qkvb  = (__bf16*)alloc((size_t)Mrows * 3 * Ee * 2);        // 24 MiB
    __bf16* Sb    = (__bf16*)alloc((size_t)Bb * Hh * Tt * Tt * 2);     // 64 MiB
    __bf16* ctxb  = (__bf16*)alloc((size_t)Mrows * Ee * 2);            //  8 MiB
    float*  x1    = (float*) alloc((size_t)Mrows * Ee * 4);            // 16 MiB
    __bf16* x1b   = (__bf16*)alloc((size_t)Mrows * Ee * 2);            //  8 MiB
    __bf16* hidb  = (__bf16*)alloc((size_t)Mrows * HIDh * 2);          // 32 MiB

    const long E3 = 3 * Ee;

    // --- fp32 -> bf16 conversions ---
    cvt_f32_bf16<<<2048, 256, 0, stream>>>(x,     xb,    (long)Mrows * Ee);
    cvt_f32_bf16<<<2048, 256, 0, stream>>>(in_w,  inwb,  (long)3 * Ee * Ee);
    cvt_f32_bf16<<<2048, 256, 0, stream>>>(out_w, outwb, (long)Ee * Ee);
    cvt_f32_bf16<<<2048, 256, 0, stream>>>(w1,    w1b,   (long)HIDh * Ee);
    cvt_f32_bf16<<<2048, 256, 0, stream>>>(w2,    w2b,   (long)Ee * HIDh);

    // --- QKV projection: qkvb = bf16(xb @ in_proj_w^T + in_proj_b) ---
    gemm_bf16_wmma<true, F_BIAS | F_WB>
        <<<dim3(E3 / BLK_N, Mrows / BLK_M, 1), 256, 0, stream>>>(
        xb, Ee, 0, 0, inwb, Ee, 0, 0, in_b, nullptr,
        nullptr, qkvb, E3, 0, 0, Mrows, (int)E3, Ee, 1, 1.f, nullptr, 0);

    // --- attention logits (scaled, causal+length masked), per (b,h) ---
    gemm_bf16_wmma<true, F_MASK | F_WF>
        <<<dim3(Tt / BLK_N, Tt / BLK_M, Bb * Hh), 256, 0, stream>>>(
        qkvb,      E3, (long)Tt * E3, DHd,                // Q slice
        qkvb + Ee, E3, (long)Tt * E3, DHd,                // K slice
        nullptr, nullptr,
        scores, nullptr, Tt, (long)Hh * Tt * Tt, (long)Tt * Tt,
        Tt, Tt, DHd, Hh, 1.f / sqrtf((float)DHd), oh, Tt);

    // --- softmax over rows of scores; bf16 copy into Sb ---
    softmax_rows<<<Bb * Hh * Tt, 256, 0, stream>>>(scores, Sb, Tt);

    // --- ctx = scores @ V (V is Kd x N row-major inside qkv), per (b,h) ---
    gemm_bf16_wmma<false, F_WB>
        <<<dim3(DHd / BLK_N, Tt / BLK_M, Bb * Hh), 256, 0, stream>>>(
        Sb,            Tt, (long)Hh * Tt * Tt, (long)Tt * Tt,
        qkvb + 2 * Ee, E3, (long)Tt * E3, DHd,
        nullptr, nullptr,
        nullptr, ctxb, Ee, (long)Tt * Ee, DHd,
        Tt, DHd, Tt, Hh, 1.f, nullptr, 0);

    // --- out projection + bias + residual(x): x1 fp32 ---
    gemm_bf16_wmma<true, F_BIAS | F_RESID | F_WF>
        <<<dim3(Ee / BLK_N, Mrows / BLK_M, 1), 256, 0, stream>>>(
        ctxb, Ee, 0, 0, outwb, Ee, 0, 0, out_b, x,
        x1, nullptr, Ee, 0, 0, Mrows, Ee, Ee, 1, 1.f, nullptr, 0);

    // --- masked BN #1 (in-place on x1, bf16 copy for MLP) ---
    bn_feature<<<Ee, 256, 0, stream>>>(x1, oh, ag, abt, x1b, Mrows, Ee);

    // --- MLP layer 1: hidb = bf16(relu(x1b @ w1^T + b1)) ---
    gemm_bf16_wmma<true, F_BIAS | F_RELU | F_WB>
        <<<dim3(HIDh / BLK_N, Mrows / BLK_M, 1), 256, 0, stream>>>(
        x1b, Ee, 0, 0, w1b, Ee, 0, 0, b1, nullptr,
        nullptr, hidb, HIDh, 0, 0, Mrows, HIDh, Ee, 1, 1.f, nullptr, 0);

    // --- MLP layer 2 + bias + residual(x1) -> x2 region of d_out ---
    gemm_bf16_wmma<true, F_BIAS | F_RESID | F_WF>
        <<<dim3(Ee / BLK_N, Mrows / BLK_M, 1), 256, 0, stream>>>(
        hidb, HIDh, 0, 0, w2b, HIDh, 0, 0, b2, x1,
        x2out, nullptr, Ee, 0, 0, Mrows, Ee, HIDh, 1, 1.f, nullptr, 0);

    // --- masked BN #2 (in-place on d_out x2 region) ---
    bn_feature<<<Ee, 256, 0, stream>>>(x2out, oh, mg, mbt, nullptr, Mrows, Ee);
}
